// MultiHeadedAttention_16183436771757
// MI455X (gfx1250) — matrix operand order repaired and checked by reading, not compiled, not run on hardware
//
#include <hip/hip_runtime.h>
#include <hip/hip_bf16.h>

typedef __attribute__((ext_vector_type(16))) _Float16 v16h;
typedef __attribute__((ext_vector_type(8)))  float    v8f;

// ---------------------------------------------------------------------------
// WMMA helper: D = A(16x32 f16) * B(32x16 f16) + C(16x16 f32)
// ---------------------------------------------------------------------------
__device__ __forceinline__ v8f wmma_f16(v16h a, v16h b, v8f c) {
  return __builtin_amdgcn_wmma_f32_16x16x32_f16(false, a, false, b, (short)0, c,
                                                false, false);
}

__device__ __forceinline__ v8f vzero8() {
  v8f z;
#pragma unroll
  for (int i = 0; i < 8; ++i) z[i] = 0.0f;
  return z;
}

// ---------------------------------------------------------------------------
// Fragment loaders (f16 source, row-major, K contiguous).
// A operand (16x32): lane L -> row (L&15); 16 contiguous K-halves at offset
//   0 (lanes 0-15) or 16 (lanes 16-31).
// B operand (32x16): lane L -> col (L&15); 16 contiguous K-halves at offset
//   0 (lanes 0-15) or 16 (lanes 16-31).
// `p` points to row_start + k0; `hi` = (lane >= 16).
// ---------------------------------------------------------------------------
__device__ __forceinline__ v16h loadA_f16(const _Float16* p, bool hi) {
  p += hi ? 16 : 0;
  v16h r;
  ((float4*)&r)[0] = *((const float4*)p);
  ((float4*)&r)[1] = *((const float4*)(p + 8));
  return r;
}

__device__ __forceinline__ v16h loadB_f16(const _Float16* p, bool hi) {
  p += hi ? 16 : 0;
  v16h r;
  ((float4*)&r)[0] = ((const float4*)p)[0];
  ((float4*)&r)[1] = ((const float4*)p)[1];
  return r;
}

// ---------------------------------------------------------------------------
// One-time f32 -> f16 conversion (n must be a multiple of 8).
// ---------------------------------------------------------------------------
__global__ __launch_bounds__(256) void cvt_f32_f16(
    const float* __restrict__ in, _Float16* __restrict__ out, int n) {
  const int i = (blockIdx.x * 256 + threadIdx.x) * 8;
  if (i >= n) return;
  const float4 a = ((const float4*)(in + i))[0];
  const float4 b = ((const float4*)(in + i))[1];
  v16h tmp;  // use low 8 halves only
  _Float16* h = (_Float16*)&tmp;
  h[0] = (_Float16)a.x; h[1] = (_Float16)a.y;
  h[2] = (_Float16)a.z; h[3] = (_Float16)a.w;
  h[4] = (_Float16)b.x; h[5] = (_Float16)b.y;
  h[6] = (_Float16)b.z; h[7] = (_Float16)b.w;
  *((float4*)(out + i)) = ((float4*)&tmp)[0];
}

// ---------------------------------------------------------------------------
// GEMM (NT): Y[r,c] = sum_k A[r,k] * W[c,k] + bias[c], all-f16 operands.
// Block = 256 threads = 8 waves; wave -> 32 rows x 64 cols; K = 1024.
// Grid: (cols/64, rows/256).  OutT = _Float16 (proj) or float (final).
// ---------------------------------------------------------------------------
template <typename OutT>
__global__ __launch_bounds__(256) void gemm_nt_kernel(
    const _Float16* __restrict__ A, const _Float16* __restrict__ W,
    const float* __restrict__ bias, OutT* __restrict__ Y) {
  const int K = 1024, NC = 1024;
  const int lane = threadIdx.x & 31;
  const int wave = threadIdx.x >> 5;
  const bool hi  = lane >= 16;
  const int l15  = lane & 15;
  const int row0 = blockIdx.y * 256 + wave * 32;
  const int col0 = blockIdx.x * 64;

  const _Float16* arow0 = A + (size_t)(row0 + l15) * K;
  const _Float16* arow1 = A + (size_t)(row0 + 16 + l15) * K;
  const _Float16* wrow  = W + (size_t)(col0 + l15) * K;

  v8f acc0[4], acc1[4];
#pragma unroll
  for (int t = 0; t < 4; ++t) { acc0[t] = vzero8(); acc1[t] = vzero8(); }

  for (int k0 = 0; k0 < K; k0 += 32) {
    const v16h a0 = loadA_f16(arow0 + k0, hi);
    const v16h a1 = loadA_f16(arow1 + k0, hi);
#pragma unroll
    for (int t = 0; t < 4; ++t) {
      const v16h b = loadB_f16(wrow + (size_t)t * 16 * K + k0, hi);
      acc0[t] = wmma_f16(a0, b, acc0[t]);
      acc1[t] = wmma_f16(a1, b, acc1[t]);
    }
  }

  const int rb = row0 + (hi ? 8 : 0);
#pragma unroll
  for (int t = 0; t < 4; ++t) {
    const int col = col0 + t * 16 + l15;
    const float bv = bias[col];
#pragma unroll
    for (int j = 0; j < 8; ++j) {
      Y[(size_t)(rb + j) * NC + col]      = (OutT)(acc0[t][j] + bv);
      Y[(size_t)(rb + 16 + j) * NC + col] = (OutT)(acc1[t][j] + bv);
    }
  }
}

// ---------------------------------------------------------------------------
// Tiled f16 transpose: in[R][C] -> out[C][R]. Grid (C/32, R/32), block (32,8).
// ---------------------------------------------------------------------------
__global__ void transpose_f16(const _Float16* __restrict__ in,
                              _Float16* __restrict__ out, int R, int C) {
  __shared__ _Float16 tile[32][33];
  const int cb = blockIdx.x * 32;
  const int rb = blockIdx.y * 32;
  const int tx = threadIdx.x;
  const int ty = threadIdx.y;
#pragma unroll
  for (int i = 0; i < 32; i += 8)
    tile[ty + i][tx] = in[(size_t)(rb + ty + i) * C + (cb + tx)];
  __syncthreads();
#pragma unroll
  for (int i = 0; i < 32; i += 8)
    out[(size_t)(cb + ty + i) * R + (rb + tx)] = tile[tx][ty + i];
}

// ---------------------------------------------------------------------------
// Flash attention, all-WMMA, transposed formulation.
// One wave handles 32 queries (two groups of 16) of one (batch-block, head):
//   S^T tile = K_tile(A) x Q^T(B)   (C layout: row=key, col=query=lane)
//   online softmax per query (lane); keys split across lane^16 halves
//   O^T     += V^T_tile(A) x P^T(B) (P^T from S^T regs + packed xor-16 swap)
// K/V fragments are loaded once per 32-key step and reused by both query
// groups (16 WMMAs per 16 b128 loads). Writes Attn^T [E x N] coalesced.
// ---------------------------------------------------------------------------
__global__ __launch_bounds__(256) void attn_kernel(
    const _Float16* __restrict__ Qh, const _Float16* __restrict__ Kh,
    const _Float16* __restrict__ VhT, _Float16* __restrict__ AttnT) {
  const int E = 1024, NR = 8192, SEQ = 2048;
  const int lane = threadIdx.x & 31;
  const int gw   = blockIdx.x * 8 + (threadIdx.x >> 5);
  const int qt   = gw & 63;           // 64 q-tiles of 32 queries = 2048
  const int hh   = (gw >> 6) & 15;    // head
  const int bb   = gw >> 10;          // batch block
  const int r0   = bb * SEQ;
  const int kcol = hh * 64;
  const bool hi  = lane >= 16;
  const int l15  = lane & 15;

  // Two query groups of 16 (B operands, kept resident)
  v16h qb[2][2];
#pragma unroll
  for (int g = 0; g < 2; ++g) {
    const int qrow = r0 + qt * 32 + g * 16 + l15;
    qb[g][0] = loadB_f16(Qh + (size_t)qrow * E + kcol, hi);
    qb[g][1] = loadB_f16(Qh + (size_t)qrow * E + kcol + 32, hi);
  }

  v8f o[2][4];
#pragma unroll
  for (int g = 0; g < 2; ++g)
#pragma unroll
    for (int t = 0; t < 4; ++t) o[g][t] = vzero8();
  float m[2] = {-3.0e38f, -3.0e38f};
  float l[2] = {0.0f, 0.0f};

  for (int kk = 0; kk < SEQ; kk += 32) {
    // K fragments for 32 keys (A operands), shared by both query groups
    const _Float16* kp_lo = Kh + (size_t)(r0 + kk + l15) * E + kcol;
    const _Float16* kp_hi = kp_lo + (size_t)16 * E;
    const v16h ka0 = loadA_f16(kp_lo, hi);
    const v16h ka1 = loadA_f16(kp_lo + 32, hi);
    const v16h ka2 = loadA_f16(kp_hi, hi);
    const v16h ka3 = loadA_f16(kp_hi + 32, hi);

    v16h pfr[2];
#pragma unroll
    for (int g = 0; g < 2; ++g) {
      v8f s_lo = vzero8(), s_hi = vzero8();
      s_lo = wmma_f16(ka0, qb[g][0], s_lo);
      s_lo = wmma_f16(ka1, qb[g][1], s_lo);
      s_hi = wmma_f16(ka2, qb[g][0], s_hi);
      s_hi = wmma_f16(ka3, qb[g][1], s_hi);

      float plo[8], phi[8];
      float mloc = -3.0e38f;
#pragma unroll
      for (int j = 0; j < 8; ++j) {
        plo[j] = s_lo[j] * 0.125f;          // 1/sqrt(64)
        phi[j] = s_hi[j] * 0.125f;
        mloc = fmaxf(mloc, fmaxf(plo[j], phi[j]));
      }
      mloc = fmaxf(mloc, __shfl_xor(mloc, 16, 32));
      const float mnew  = fmaxf(m[g], mloc);
      const float scale = __expf(m[g] - mnew);
      float rsum = 0.0f;
#pragma unroll
      for (int j = 0; j < 8; ++j) {
        plo[j] = __expf(plo[j] - mnew);
        phi[j] = __expf(phi[j] - mnew);
        rsum += plo[j] + phi[j];
      }
      rsum += __shfl_xor(rsum, 16, 32);
      l[g] = l[g] * scale + rsum;
      m[g] = mnew;

      // P^T B operand (32 keys x 16 queries). Lane q owns keys
      // {kk..kk+7, kk+16..kk+23}; lane q+16 owns {kk+8..15, kk+24..31}.
      // To match V^T A (contiguous k), B needs lane<16 : keys 0..15, lane>=16 : keys 16..31.
      // Pack (p_lo, p_hi) as f16x2 and exchange once per pair.
      _Float16* pf = (_Float16*)&pfr[g];
#pragma unroll
      for (int j = 0; j < 8; ++j) {
        union { unsigned int u; _Float16 h[2]; } own, oth;
        own.h[0] = (_Float16)plo[j];
        own.h[1] = (_Float16)phi[j];
        oth.u = (unsigned int)__shfl_xor((int)own.u, 16, 32);
        pf[j]     = hi ? oth.h[1] : own.h[0];
        pf[8 + j] = hi ? own.h[1] : oth.h[0];
      }

#pragma unroll
      for (int t = 0; t < 4; ++t)
#pragma unroll
        for (int j = 0; j < 8; ++j) o[g][t][j] *= scale;
    }

    // V^T fragments (A operands), shared by both query groups
    const _Float16* vbase = VhT + (size_t)(kcol + l15) * NR + (r0 + kk);
#pragma unroll
    for (int t = 0; t < 4; ++t) {
      const v16h vf = loadA_f16(vbase + (size_t)t * 16 * NR, hi);
      o[0][t] = wmma_f16(vf, pfr[0], o[0][t]);
      o[1][t] = wmma_f16(vf, pfr[1], o[1][t]);
    }
  }

  const int ebase = kcol + (hi ? 8 : 0);
#pragma unroll
  for (int g = 0; g < 2; ++g) {
    const float inv = 1.0f / l[g];
    const int qc = r0 + qt * 32 + g * 16 + l15;
#pragma unroll
    for (int t = 0; t < 4; ++t)
#pragma unroll
      for (int j = 0; j < 8; ++j)
        AttnT[(size_t)(ebase + t * 16 + j) * NR + qc] =
            (_Float16)(o[g][t][j] * inv);
  }
}

// ---------------------------------------------------------------------------
extern "C" void kernel_launch(void* const* d_in, const int* in_sizes, int n_in,
                              void* d_out, int out_size, void* d_ws,
                              size_t ws_size, hipStream_t stream) {
  (void)in_sizes; (void)n_in; (void)out_size; (void)ws_size;

  const float* q  = (const float*)d_in[0];
  const float* k  = (const float*)d_in[1];
  const float* v  = (const float*)d_in[2];
  const float* Wq = (const float*)d_in[3];
  const float* bq = (const float*)d_in[4];
  const float* Wk = (const float*)d_in[5];
  const float* bk = (const float*)d_in[6];
  const float* Wv = (const float*)d_in[7];
  const float* bvp= (const float*)d_in[8];
  const float* Wo = (const float*)d_in[9];
  const float* bo = (const float*)d_in[10];
  float* out = (float*)d_out;

  const size_t NE = (size_t)8192 * 1024;   // activations  [N x E]
  const size_t EE = (size_t)1024 * 1024;   // weights      [E x E]
  _Float16* Qh    = (_Float16*)d_ws;       // projected Q (f16)
  _Float16* Kh    = Qh + 1 * NE;
  _Float16* Vh    = Qh + 2 * NE;
  _Float16* VhT   = Qh + 3 * NE;           // V transposed [E x N]
  _Float16* AttnT = Qh + 4 * NE;           // attn output  [E x N]
  _Float16* Attn  = Qh + 5 * NE;           // attn output  [N x E]
  _Float16* Xq    = Qh + 6 * NE;           // f16 copies of inputs
  _Float16* Xk    = Qh + 7 * NE;
  _Float16* Xv    = Qh + 8 * NE;
  _Float16* Wqh   = Qh + 9 * NE;           // f16 copies of weights
  _Float16* Wkh   = Wqh + 1 * EE;
  _Float16* Wvh   = Wqh + 2 * EE;
  _Float16* Woh   = Wqh + 3 * EE;

  // One-time conversions (f32 -> f16)
  const int nAct = (int)NE, nW = (int)EE;
  cvt_f32_f16<<<dim3(nAct / 8 / 256), dim3(256), 0, stream>>>(q, Xq, nAct);
  cvt_f32_f16<<<dim3(nAct / 8 / 256), dim3(256), 0, stream>>>(k, Xk, nAct);
  cvt_f32_f16<<<dim3(nAct / 8 / 256), dim3(256), 0, stream>>>(v, Xv, nAct);
  cvt_f32_f16<<<dim3(nW / 8 / 256), dim3(256), 0, stream>>>(Wq, Wqh, nW);
  cvt_f32_f16<<<dim3(nW / 8 / 256), dim3(256), 0, stream>>>(Wk, Wkh, nW);
  cvt_f32_f16<<<dim3(nW / 8 / 256), dim3(256), 0, stream>>>(Wv, Wvh, nW);
  cvt_f32_f16<<<dim3(nW / 8 / 256), dim3(256), 0, stream>>>(Wo, Woh, nW);

  // Projections: Q/K/V = X @ W^T + b   (f16 out)
  const dim3 gGemm(16, 32), bGemm(256);
  gemm_nt_kernel<_Float16><<<gGemm, bGemm, 0, stream>>>(Xq, Wqh, bq, Qh);
  gemm_nt_kernel<_Float16><<<gGemm, bGemm, 0, stream>>>(Xk, Wkh, bk, Kh);
  gemm_nt_kernel<_Float16><<<gGemm, bGemm, 0, stream>>>(Xv, Wvh, bvp, Vh);

  // V -> V^T for contiguous V^T A-operand loads in attention
  transpose_f16<<<dim3(1024 / 32, 8192 / 32), dim3(32, 8), 0, stream>>>(
      Vh, VhT, 8192, 1024);

  // Flash attention: 4 row-blocks x 16 heads x 64 q-tiles = 4096 waves
  attn_kernel<<<dim3(512), dim3(256), 0, stream>>>(Qh, Kh, VhT, AttnT);

  // Attn^T -> Attn for the final GEMM's A operand
  transpose_f16<<<dim3(8192 / 32, 1024 / 32), dim3(32, 8), 0, stream>>>(
      AttnT, Attn, 1024, 8192);

  // Out = Attn @ Wo^T + bo   (f32 out)
  gemm_nt_kernel<float><<<gGemm, bGemm, 0, stream>>>(Attn, Woh, bo, out);
}
